// SwitchFeedForward_9955734192320
// MI455X (gfx1250) — matrix operand order repaired and checked by reading, not compiled, not run on hardware
//
#include <hip/hip_runtime.h>
#include <math.h>

// Problem constants (B=2, T=2048, C=1024, H=4096, E=8, K=2)
#define S_TOK 4096
#define C_DIM 1024
#define H_DIM 4096
#define E_NUM 8
#define TOPK  2
#define CAP   2048   // per-expert token capacity (avg load = 1024 for top-2/8)

// GEMM tiling: 128x128 tile, K-step 32 (= WMMA bf16 K), 8 waves (4x2 wave grid)
#define BM 128
#define BN 128
#define BK 32
#define LDA 40   // ushorts per LDS row: BK + 8 pad (80B, keeps 16B alignment)
#define LDB 40

typedef __attribute__((ext_vector_type(16))) __bf16 v16bf;
typedef __attribute__((ext_vector_type(8)))  float  v8f;

union Frag { uint4 q[2]; v16bf v; };

// fp32 -> bf16 round-to-nearest-even (bit pattern)
__device__ __forceinline__ unsigned short f2bf(float f) {
  unsigned int u = __float_as_uint(f);
  u += 0x7FFFu + ((u >> 16) & 1u);
  return (unsigned short)(u >> 16);
}

__device__ __forceinline__ float gelu_exact(float x) {
  return 0.5f * x * (1.0f + erff(x * 0.70710678118654752f));
}

// Generic pointer -> LDS byte offset (generic LDS addr carries offset in [31:0])
__device__ __forceinline__ unsigned lds_off_of(const void* p) {
  return (unsigned)(unsigned long long)p;
}

// CDNA5 async global->LDS copy, 16B per lane (ASYNCcnt tracked)
__device__ __forceinline__ void async_b128(unsigned lds, const void* g) {
  asm volatile("global_load_async_to_lds_b128 %0, %1, off"
               :: "v"(lds), "v"(g) : "memory");
}
__device__ __forceinline__ void async_wait0() {
  asm volatile("s_wait_asynccnt 0x0" ::: "memory");
}

// ---------------------------------------------------------------------------
// Kernel 0: zero output accumulator + expert counters
// ---------------------------------------------------------------------------
__global__ __launch_bounds__(256) void moe_zero(float* __restrict__ out,
                                                int* __restrict__ counts) {
  const size_t i = ((size_t)blockIdx.x * 256 + threadIdx.x) * 4;
  *(float4*)(out + i) = make_float4(0.f, 0.f, 0.f, 0.f);
  if (blockIdx.x == 0 && threadIdx.x < E_NUM) counts[threadIdx.x] = 0;
}

// ---------------------------------------------------------------------------
// Kernel 1: one-shot activation convert: x fp32 -> bf16 (row-major, A-operand
// major). 8 threads*... each thread converts 8 elements.
// ---------------------------------------------------------------------------
__global__ __launch_bounds__(256) void cvt_x(const float* __restrict__ x,
                                             unsigned short* __restrict__ xbf) {
  const size_t i = ((size_t)blockIdx.x * 256 + threadIdx.x) * 8;
  const float4 a = *(const float4*)(x + i);
  const float4 b = *(const float4*)(x + i + 4);
  uint4 q;
  q.x = (unsigned)f2bf(a.x) | ((unsigned)f2bf(a.y) << 16);
  q.y = (unsigned)f2bf(a.z) | ((unsigned)f2bf(a.w) << 16);
  q.z = (unsigned)f2bf(b.x) | ((unsigned)f2bf(b.y) << 16);
  q.w = (unsigned)f2bf(b.z) | ((unsigned)f2bf(b.w) << 16);
  *(uint4*)(xbf + i) = q;
}

// ---------------------------------------------------------------------------
// Kernel 2: one-shot weight transform: fp32 [e][k][n] -> bf16 [e][n][k].
// Makes weights L2-resident (134 MB bf16 < 192 MB L2) and B-operand-major,
// so GEMM loops become pure async bf16 copies (no in-loop cvt/transpose).
// ---------------------------------------------------------------------------
__global__ __launch_bounds__(256) void transpose_cvt(
    const float* __restrict__ src, unsigned short* __restrict__ dst,
    int K, int N) {
  __shared__ unsigned short tile[32][33];
  const int e  = blockIdx.z;
  const int n0 = blockIdx.x * 32;
  const int k0 = blockIdx.y * 32;
  const float* S = src + (size_t)e * K * N;
  unsigned short* D = dst + (size_t)e * N * K;
  const int tx = threadIdx.x & 31;
  const int ty = threadIdx.x >> 5;
#pragma unroll
  for (int r = ty; r < 32; r += 8)
    tile[r][tx] = f2bf(S[(size_t)(k0 + r) * N + n0 + tx]);
  __syncthreads();
#pragma unroll
  for (int r = ty; r < 32; r += 8)
    D[(size_t)(n0 + r) * K + k0 + tx] = tile[tx][r];
}

// ---------------------------------------------------------------------------
// Kernel 3: router. One wave32 per token: logits = x[t] @ Wg, softmax, top-2.
// ---------------------------------------------------------------------------
__global__ __launch_bounds__(256) void moe_router(
    const float* __restrict__ x, const float* __restrict__ Wg,
    float* __restrict__ logits_out,
    int* __restrict__ counts, int* __restrict__ tokId,
    float* __restrict__ tokW) {
  const int wave = threadIdx.x >> 5;
  const int lane = threadIdx.x & 31;
  const int t = blockIdx.x * 8 + wave;
  const float* xr = x + (size_t)t * C_DIM;

  float acc[E_NUM];
#pragma unroll
  for (int e = 0; e < E_NUM; ++e) acc[e] = 0.f;

  for (int c = lane; c < C_DIM; c += 32) {
    const float xv = xr[c];
    const float4 g0 = *(const float4*)(Wg + (size_t)c * E_NUM);
    const float4 g1 = *(const float4*)(Wg + (size_t)c * E_NUM + 4);
    acc[0] += xv * g0.x; acc[1] += xv * g0.y;
    acc[2] += xv * g0.z; acc[3] += xv * g0.w;
    acc[4] += xv * g1.x; acc[5] += xv * g1.y;
    acc[6] += xv * g1.z; acc[7] += xv * g1.w;
  }
#pragma unroll
  for (int e = 0; e < E_NUM; ++e) {
#pragma unroll
    for (int off = 16; off >= 1; off >>= 1)
      acc[e] += __shfl_xor(acc[e], off, 32);
  }

  float m = acc[0];
#pragma unroll
  for (int e = 1; e < E_NUM; ++e) m = fmaxf(m, acc[e]);
  float p[E_NUM]; float sum = 0.f;
#pragma unroll
  for (int e = 0; e < E_NUM; ++e) { p[e] = expf(acc[e] - m); sum += p[e]; }
  const float inv = 1.0f / sum;
#pragma unroll
  for (int e = 0; e < E_NUM; ++e) p[e] *= inv;

  int e0 = 0; float v0 = p[0];
#pragma unroll
  for (int e = 1; e < E_NUM; ++e) if (p[e] > v0) { v0 = p[e]; e0 = e; }
  int e1 = (e0 == 0) ? 1 : 0; float v1 = p[e1];
#pragma unroll
  for (int e = 0; e < E_NUM; ++e)
    if (e != e0 && p[e] > v1) { v1 = p[e]; e1 = e; }

  if (lane == 0) {
    float* lo = logits_out + (size_t)t * E_NUM;
    *(float4*)(lo)     = make_float4(acc[0], acc[1], acc[2], acc[3]);
    *(float4*)(lo + 4) = make_float4(acc[4], acc[5], acc[6], acc[7]);
    int s0 = atomicAdd(&counts[e0], 1);
    if (s0 < CAP) { tokId[e0 * CAP + s0] = t; tokW[e0 * CAP + s0] = v0; }
    int s1 = atomicAdd(&counts[e1], 1);
    if (s1 < CAP) { tokId[e1 * CAP + s1] = t; tokW[e1 * CAP + s1] = v1; }
  }
}

// ---------------------------------------------------------------------------
// Kernel 4: FFN1 grouped GEMM. Hbuf[e,slot,:] = bf16(gelu(x@W1[e]) * w)
// A: gathered xbf rows via async copy. B: async bf16 copy from W1t.
// Double-buffered LDS, one barrier per K-step, zero in-loop conversions.
// ---------------------------------------------------------------------------
__global__ __launch_bounds__(256, 2) void moe_ffn1(
    const unsigned short* __restrict__ xbf, const unsigned short* __restrict__ W1t,
    const int* __restrict__ counts, const int* __restrict__ tokId,
    const float* __restrict__ tokW, unsigned short* __restrict__ Hbuf) {
  __shared__ alignas(16) unsigned short As[2][BM * LDA];
  __shared__ alignas(16) unsigned short Bs[2][BN * LDB];
  __shared__ int   rowTok[BM];
  __shared__ float rowW[BM];

  const int tiles_m = CAP / BM;     // 16
  const int tiles_n = H_DIM / BN;   // 32
  int bx = blockIdx.x;
  const int e = bx / (tiles_m * tiles_n);
  bx -= e * tiles_m * tiles_n;
  const int mtile = bx / tiles_n;
  const int ntile = bx - mtile * tiles_n;

  const int cnt = min(counts[e], CAP);
  const int m0 = mtile * BM;
  if (m0 >= cnt) return;

  const int tid = threadIdx.x;
  if (tid < BM) {
    const int slot = m0 + tid;
    const int valid = slot < cnt;
    rowTok[tid] = valid ? tokId[e * CAP + slot] : 0;
    rowW[tid]   = valid ? tokW[e * CAP + slot]  : 0.f;
  }
  __syncthreads();

  const int n0 = ntile * BN;
  const int lrow = tid >> 1;   // loader row (0..127)
  const int lseg = tid & 1;    // 32B half of the 64B K-slab row
  const unsigned short* asrc =
      xbf + (size_t)rowTok[lrow] * C_DIM + lseg * 16;   // gathered token row
  const unsigned short* bsrc =
      W1t + ((size_t)e * H_DIM + n0 + lrow) * C_DIM + lseg * 16;
  unsigned aoff[2], boff[2];
#pragma unroll
  for (int b = 0; b < 2; ++b) {
    aoff[b] = lds_off_of(&As[b][lrow * LDA + lseg * 16]);
    boff[b] = lds_off_of(&Bs[b][lrow * LDB + lseg * 16]);
  }

  const int wave = tid >> 5, lane = tid & 31;
  const int waveM = wave >> 1, waveN = wave & 1;
  const int lhal = lane >> 4, lmod = lane & 15;

  v8f acc[2][4];
#pragma unroll
  for (int mt = 0; mt < 2; ++mt)
#pragma unroll
    for (int nt = 0; nt < 4; ++nt) acc[mt][nt] = (v8f)(0.0f);

  // ---- prologue: tile 0 (pure async) ----
  async_b128(aoff[0], asrc);
  async_b128(aoff[0] + 16, asrc + 8);
  async_b128(boff[0], bsrc);
  async_b128(boff[0] + 16, bsrc + 8);
  async_wait0();
  __syncthreads();

  int buf = 0;
  for (int k0 = 0; k0 < C_DIM; k0 += BK) {
    const bool more = (k0 + BK) < C_DIM;
    if (more) {
      async_b128(aoff[buf ^ 1], asrc + k0 + BK);
      async_b128(aoff[buf ^ 1] + 16, asrc + k0 + BK + 8);
      async_b128(boff[buf ^ 1], bsrc + k0 + BK);
      async_b128(boff[buf ^ 1] + 16, bsrc + k0 + BK + 8);
    }
    Frag a[2], b[4];
#pragma unroll
    for (int mt = 0; mt < 2; ++mt) {
      const unsigned short* ap =
          &As[buf][(waveM * 32 + mt * 16 + lmod) * LDA + lhal * 8];
      a[mt].q[0] = *(const uint4*)(ap);
      a[mt].q[1] = *(const uint4*)(ap + 16);
    }
#pragma unroll
    for (int nt = 0; nt < 4; ++nt) {
      const unsigned short* bp =
          &Bs[buf][(waveN * 64 + nt * 16 + lmod) * LDB + lhal * 8];
      b[nt].q[0] = *(const uint4*)(bp);
      b[nt].q[1] = *(const uint4*)(bp + 16);
    }
#pragma unroll
    for (int mt = 0; mt < 2; ++mt)
#pragma unroll
      for (int nt = 0; nt < 4; ++nt)
        acc[mt][nt] = __builtin_amdgcn_wmma_f32_16x16x32_bf16(
            false, a[mt].v, false, b[nt].v, (short)0, acc[mt][nt], false, false);
    if (more) {
      async_wait0();
      __syncthreads();
    }
    buf ^= 1;
  }

  // ---- epilogue: gelu * combine-weight, bf16 store to Hbuf ----
  const size_t hbase = (size_t)e * CAP * H_DIM;
#pragma unroll
  for (int mt = 0; mt < 2; ++mt) {
#pragma unroll
    for (int r = 0; r < 8; ++r) {
      const int mw = waveM * 32 + mt * 16 + r + lhal * 8;
      const int slot = m0 + mw;
      if (slot < cnt) {
        const float w = rowW[mw];
#pragma unroll
        for (int nt = 0; nt < 4; ++nt) {
          const int hcol = n0 + waveN * 64 + nt * 16 + lmod;
          Hbuf[hbase + (size_t)slot * H_DIM + hcol] =
              f2bf(gelu_exact(acc[mt][nt][r]) * w);
        }
      }
    }
  }
}

// ---------------------------------------------------------------------------
// Kernel 5: FFN2 grouped GEMM. out[tok] += Hbuf[e,slot,:] @ W2[e]
// Both A (Hbuf bf16) and B (W2t bf16) tiles stream via async LDS copies.
// Exactly 2 commutative f32 atomic adds per output element -> deterministic.
// ---------------------------------------------------------------------------
__global__ __launch_bounds__(256, 2) void moe_ffn2(
    const unsigned short* __restrict__ Hbuf, const unsigned short* __restrict__ W2t,
    const int* __restrict__ counts, const int* __restrict__ tokId,
    float* __restrict__ out) {
  __shared__ alignas(16) unsigned short As[2][BM * LDA];
  __shared__ alignas(16) unsigned short Bs[2][BN * LDB];
  __shared__ int rowTok[BM];

  const int tiles_m = CAP / BM;     // 16
  const int tiles_n = C_DIM / BN;   // 8
  int bx = blockIdx.x;
  const int e = bx / (tiles_m * tiles_n);
  bx -= e * tiles_m * tiles_n;
  const int mtile = bx / tiles_n;
  const int ntile = bx - mtile * tiles_n;

  const int cnt = min(counts[e], CAP);
  const int m0 = mtile * BM;
  if (m0 >= cnt) return;

  const int tid = threadIdx.x;
  if (tid < BM) {
    const int slot = m0 + tid;
    rowTok[tid] = (slot < cnt) ? tokId[e * CAP + slot] : 0;
  }

  const int n0 = ntile * BN;
  const int lrow = tid >> 1, lseg = tid & 1;
  const unsigned short* asrc =
      Hbuf + ((size_t)e * CAP + m0 + lrow) * H_DIM + lseg * 16;
  const unsigned short* bsrc =
      W2t + ((size_t)e * C_DIM + n0 + lrow) * H_DIM + lseg * 16;
  unsigned aoff[2], boff[2];
#pragma unroll
  for (int b = 0; b < 2; ++b) {
    aoff[b] = lds_off_of(&As[b][lrow * LDA + lseg * 16]);
    boff[b] = lds_off_of(&Bs[b][lrow * LDB + lseg * 16]);
  }

  const int wave = tid >> 5, lane = tid & 31;
  const int waveM = wave >> 1, waveN = wave & 1;
  const int lhal = lane >> 4, lmod = lane & 15;

  v8f acc[2][4];
#pragma unroll
  for (int mt = 0; mt < 2; ++mt)
#pragma unroll
    for (int nt = 0; nt < 4; ++nt) acc[mt][nt] = (v8f)(0.0f);

  // ---- prologue: tile 0 (pure async) ----
  async_b128(aoff[0], asrc);
  async_b128(aoff[0] + 16, asrc + 8);
  async_b128(boff[0], bsrc);
  async_b128(boff[0] + 16, bsrc + 8);
  async_wait0();
  __syncthreads();

  int buf = 0;
  for (int k0 = 0; k0 < H_DIM; k0 += BK) {
    const bool more = (k0 + BK) < H_DIM;
    if (more) {
      async_b128(aoff[buf ^ 1], asrc + k0 + BK);
      async_b128(aoff[buf ^ 1] + 16, asrc + k0 + BK + 8);
      async_b128(boff[buf ^ 1], bsrc + k0 + BK);
      async_b128(boff[buf ^ 1] + 16, bsrc + k0 + BK + 8);
    }
    Frag a[2], b[4];
#pragma unroll
    for (int mt = 0; mt < 2; ++mt) {
      const unsigned short* ap =
          &As[buf][(waveM * 32 + mt * 16 + lmod) * LDA + lhal * 8];
      a[mt].q[0] = *(const uint4*)(ap);
      a[mt].q[1] = *(const uint4*)(ap + 16);
    }
#pragma unroll
    for (int nt = 0; nt < 4; ++nt) {
      const unsigned short* bp =
          &Bs[buf][(waveN * 64 + nt * 16 + lmod) * LDB + lhal * 8];
      b[nt].q[0] = *(const uint4*)(bp);
      b[nt].q[1] = *(const uint4*)(bp + 16);
    }
#pragma unroll
    for (int mt = 0; mt < 2; ++mt)
#pragma unroll
      for (int nt = 0; nt < 4; ++nt)
        acc[mt][nt] = __builtin_amdgcn_wmma_f32_16x16x32_bf16(
            false, a[mt].v, false, b[nt].v, (short)0, acc[mt][nt], false, false);
    if (more) {
      async_wait0();
      __syncthreads();
    }
    buf ^= 1;
  }

  // ---- epilogue: scatter-accumulate into out (2 adds/element total) ----
#pragma unroll
  for (int mt = 0; mt < 2; ++mt) {
#pragma unroll
    for (int r = 0; r < 8; ++r) {
      const int mw = waveM * 32 + mt * 16 + r + lhal * 8;
      const int slot = m0 + mw;
      if (slot < cnt) {
        const int tok = rowTok[mw];
#pragma unroll
        for (int nt = 0; nt < 4; ++nt) {
          const int ccol = n0 + waveN * 64 + nt * 16 + lmod;
          atomicAdd(&out[(size_t)tok * C_DIM + ccol], acc[mt][nt][r]);
        }
      }
    }
  }
}

// ---------------------------------------------------------------------------
extern "C" void kernel_launch(void* const* d_in, const int* in_sizes, int n_in,
                              void* d_out, int out_size, void* d_ws, size_t ws_size,
                              hipStream_t stream) {
  (void)in_sizes; (void)n_in; (void)out_size; (void)ws_size;
  const float* x  = (const float*)d_in[0];
  const float* W1 = (const float*)d_in[1];
  const float* W2 = (const float*)d_in[2];
  const float* Wg = (const float*)d_in[3];
  float* out    = (float*)d_out;
  float* logits = out + (size_t)S_TOK * C_DIM;   // tuple tail: raw logits [S,E]

  // workspace layout (~264.1 MiB total)
  char* ws = (char*)d_ws;
  int*   counts = (int*)(ws);                       //  32 B
  int*   tokId  = (int*)(ws + 256);                 //  64 KiB
  float* tokW   = (float*)(ws + 256 + 65536);       //  64 KiB
  unsigned short* xbf = (unsigned short*)(ws + 131328);              //  8 MiB
  unsigned short* W1t = (unsigned short*)(ws + 131328 + 8388608);    // 64 MiB
  unsigned short* W2t =
      (unsigned short*)(ws + 131328 + 8388608 + 67108864);           // 64 MiB
  unsigned short* Hbuf =
      (unsigned short*)(ws + 131328 + 8388608 + 2ull * 67108864);    // 128 MiB

  moe_zero<<<(S_TOK * C_DIM) / 1024, 256, 0, stream>>>(out, counts);
  cvt_x<<<(S_TOK * C_DIM) / 2048, 256, 0, stream>>>(x, xbf);
  transpose_cvt<<<dim3(H_DIM / 32, C_DIM / 32, E_NUM), 256, 0, stream>>>(
      W1, W1t, C_DIM, H_DIM);
  transpose_cvt<<<dim3(C_DIM / 32, H_DIM / 32, E_NUM), 256, 0, stream>>>(
      W2, W2t, H_DIM, C_DIM);
  moe_router<<<S_TOK / 8, 256, 0, stream>>>(x, Wg, logits, counts, tokId, tokW);
  moe_ffn1<<<E_NUM * (CAP / BM) * (H_DIM / BN), 256, 0, stream>>>(
      xbf, W1t, counts, tokId, tokW, Hbuf);
  moe_ffn2<<<E_NUM * (CAP / BM) * (C_DIM / BN), 256, 0, stream>>>(
      Hbuf, W2t, counts, tokId, out);
}
